// ArcFaceLoss_80771154969349
// MI455X (gfx1250) — matrix operand order repaired and checked by reading, not compiled, not run on hardware
//
#include <hip/hip_runtime.h>
#include <hip/hip_bf16.h>

// ---------------- problem constants ----------------
#define BATCH   512
#define NCLS    100000
#define EMBED   512
#define MARGIN  0.5f
#define SCALE   64.0f
#define EPS     1e-7f

// ---------------- tiling ----------------
#define BT   128            // batch tile per block
#define CT   128            // class tile per block
#define KC   64             // K chunk staged per iteration
#define KCP  72             // padded LDS row stride (halves) -> conflict-free ds reads
#define NBT  (BATCH / BT)                   // 4
#define NCT  ((NCLS + CT - 1) / CT)         // 782 (last tile: 32 valid classes)
#define THREADS 512                          // 16 waves; wave tile = 32x32

typedef __attribute__((ext_vector_type(16))) _Float16 v16h;
typedef __attribute__((ext_vector_type(8)))  _Float16 v8h;
typedef __attribute__((ext_vector_type(8)))  float    v8f;

// order-preserving float <-> u32 key (for ds_max_u32 based float max)
__device__ __forceinline__ unsigned fenc(float f) {
    unsigned u = __float_as_uint(f);
    return (u & 0x80000000u) ? ~u : (u | 0x80000000u);
}
__device__ __forceinline__ float fdec(unsigned k) {
    unsigned u = (k & 0x80000000u) ? (k ^ 0x80000000u) : ~k;
    return __uint_as_float(u);
}

// =====================================================================
// Kernel 1: L2-normalize embeddings (f32) -> f16 row-major [BATCH][EMBED]
// =====================================================================
__global__ void arcface_norm_e(const float* __restrict__ emb,
                               _Float16* __restrict__ e16) {
    const int b = blockIdx.x;          // 512 blocks
    const int t = threadIdx.x;         // 128 threads, 4 floats each
    float4 v = *(const float4*)(emb + (size_t)b * EMBED + t * 4);
    float ss = v.x * v.x + v.y * v.y + v.z * v.z + v.w * v.w;
    __shared__ float red[128];
    red[t] = ss;
    __syncthreads();
    for (int s = 64; s > 0; s >>= 1) {
        if (t < s) red[t] += red[t + s];
        __syncthreads();
    }
    const float inv = 1.0f / fmaxf(sqrtf(red[0]), 1e-12f);
    _Float16* o = e16 + (size_t)b * EMBED + t * 4;
    o[0] = (_Float16)(v.x * inv);
    o[1] = (_Float16)(v.y * inv);
    o[2] = (_Float16)(v.z * inv);
    o[3] = (_Float16)(v.w * inv);
}

// =====================================================================
// Kernel 2: fused  dot(e_norm, w) / ||w||  -> margin -> per-tile
//           online-softmax partials (max, sum-exp) + target logit.
// grid = (NBT, NCT), block = 512 threads (16 waves, 32x32 tile each)
// =====================================================================
__global__ void __launch_bounds__(THREADS, 1)
arcface_gemm(const _Float16* __restrict__ e16,
             const float*    __restrict__ weight,
             const int*      __restrict__ labels,
             float* __restrict__ starget,   // [BATCH]   scaled target logit
             float* __restrict__ pmax,      // [BATCH][NCT]
             float* __restrict__ psum) {    // [BATCH][NCT]
    const int tid   = threadIdx.x;
    const int lane  = tid & 31;
    const int wid   = tid >> 5;            // 0..15
    const int hi    = lane >> 4;           // K-half selector
    const int nl    = lane & 15;           // M (for A) / N (for B) within tile
    const int Mbase = (wid & 3) * 32;      // wave's batch-row base (0,32,64,96)
    const int Nbase = (wid >> 2) * 32;     // wave's class-col base (0,32,64,96)
    const int b0    = blockIdx.x * BT;
    const int c0    = blockIdx.y * CT;

    __shared__ _Float16 wt[CT * KCP];      // staged f16 weight chunk (18 KB)
    __shared__ float    ssq4[CT * 4];      // per-class sumsq partials
    __shared__ float    invn[CT];          // 1/||w_c||
    __shared__ unsigned umax[BT];          // encoded per-row max
    __shared__ float    fsum[BT];          // per-row sum of exp

    v8f acc[2][2];
    const v8f vzero = {};
#pragma unroll
    for (int mi = 0; mi < 2; ++mi)
#pragma unroll
        for (int ni = 0; ni < 2; ++ni) acc[mi][ni] = vzero;

    // cooperative-load assignment: 4 threads per class row, 16 floats each
    const int lrow = tid >> 2;             // 0..127
    const int lqtr = tid & 3;              // 0..3  (16-float quarter of KC)
    float ssq = 0.0f;

    for (int kc = 0; kc < EMBED; kc += KC) {
        // ---- stage weight chunk: f32 global -> f16 LDS, fused sumsq ----
        {
            const int cg = c0 + lrow;
            _Float16* dp = &wt[lrow * KCP + lqtr * 16];
            if (cg < NCLS) {
                const float* wp = weight + (size_t)cg * EMBED + kc + lqtr * 16;
#pragma unroll
                for (int j = 0; j < 4; ++j) {
                    float4 w4 = *(const float4*)(wp + j * 4);
                    ssq += w4.x * w4.x + w4.y * w4.y + w4.z * w4.z + w4.w * w4.w;
                    dp[j * 4 + 0] = (_Float16)w4.x;
                    dp[j * 4 + 1] = (_Float16)w4.y;
                    dp[j * 4 + 2] = (_Float16)w4.z;
                    dp[j * 4 + 3] = (_Float16)w4.w;
                }
            } else {
#pragma unroll
                for (int j = 0; j < 16; ++j) dp[j] = (_Float16)0.0f;
            }
        }
        __syncthreads();

        // ---- 2 WMMA K-steps of 32 over this chunk ----
#pragma unroll
        for (int kk = 0; kk < KC; kk += 32) {
            // A fragments (global, row-major f16): lane nl = row M,
            // lane-half hi selects K {8*hi..8*hi+7} and {16+8*hi..}
            v16h af[2];
#pragma unroll
            for (int mi = 0; mi < 2; ++mi) {
                const _Float16* ap = e16
                    + (size_t)(b0 + Mbase + mi * 16 + nl) * EMBED
                    + (kc + kk) + hi * 8;
                v8h alo = *(const v8h*)ap;
                v8h ahi = *(const v8h*)(ap + 16);
#pragma unroll
                for (int i = 0; i < 8; ++i) { af[mi][i] = alo[i]; af[mi][i + 8] = ahi[i]; }
            }
            // B fragments (LDS, row-major f16): lane nl = column N (class),
            // hi selects K {8*hi..8*hi+7} and {16+8*hi..} of the step
            v16h bf[2];
#pragma unroll
            for (int ni = 0; ni < 2; ++ni) {
                const _Float16* bp = &wt[(Nbase + ni * 16 + nl) * KCP + kk + hi * 8];
                v8h blo = *(const v8h*)bp;
                v8h bhi = *(const v8h*)(bp + 16);
#pragma unroll
                for (int i = 0; i < 8; ++i) { bf[ni][i] = blo[i]; bf[ni][i + 8] = bhi[i]; }
            }
#pragma unroll
            for (int mi = 0; mi < 2; ++mi)
#pragma unroll
                for (int ni = 0; ni < 2; ++ni)
                    acc[mi][ni] = __builtin_amdgcn_wmma_f32_16x16x32_f16(
                        false, af[mi], false, bf[ni],
                        (short)0, acc[mi][ni], false, false);
        }
        __syncthreads();
    }

    // ---- per-class inverse norms + reduction state init ----
    ssq4[lrow * 4 + lqtr] = ssq;
    __syncthreads();
    if (tid < CT) {
        float ss = ssq4[tid * 4] + ssq4[tid * 4 + 1]
                 + ssq4[tid * 4 + 2] + ssq4[tid * 4 + 3];
        invn[tid] = 1.0f / fmaxf(sqrtf(ss), 1e-12f);
    }
    if (tid < BT) {
        umax[tid] = fenc(-INFINITY);
        fsum[tid] = 0.0f;
    }
    __syncthreads();

    // ---- epilogue pass 1: logits (margin at label), per-row max ----
    // C/D layout: VGPR r, lanes 0-15 -> M=r, N=lane; lanes 16-31 -> M=r+8
#pragma unroll
    for (int mi = 0; mi < 2; ++mi) {
#pragma unroll
        for (int r = 0; r < 8; ++r) {
            const int bl  = Mbase + mi * 16 + hi * 8 + r;
            const int bg  = b0 + bl;
            const int lbl = labels[bg];
            float mloc = -INFINITY;
#pragma unroll
            for (int ni = 0; ni < 2; ++ni) {
                const int cl = Nbase + ni * 16 + nl;
                const int cg = c0 + cl;
                float x;
                if (cg < NCLS) {
                    float cosv = acc[mi][ni][r] * invn[cl];
                    if (cg == lbl) {
                        float cc = fminf(fmaxf(cosv, -1.0f + EPS), 1.0f - EPS);
                        x = SCALE * cosf(acosf(cc) + MARGIN);
                        starget[bg] = x;        // unique writer for (b, label)
                    } else {
                        x = SCALE * cosv;
                    }
                } else {
                    x = -INFINITY;              // padded class
                }
                acc[mi][ni][r] = x;             // stash logit back in accumulator
                mloc = fmaxf(mloc, x);
            }
            atomicMax(&umax[bl], fenc(mloc));   // ds_max_u32
        }
    }
    __syncthreads();

    // ---- epilogue pass 2: sum of exp(logit - rowmax) ----
#pragma unroll
    for (int mi = 0; mi < 2; ++mi) {
#pragma unroll
        for (int r = 0; r < 8; ++r) {
            const int bl = Mbase + mi * 16 + hi * 8 + r;
            const float M = fdec(umax[bl]);
            float s = 0.0f;
#pragma unroll
            for (int ni = 0; ni < 2; ++ni)
                s += expf(acc[mi][ni][r] - M);  // exp(-inf)=0 for padding
            atomicAdd(&fsum[bl], s);            // ds_add_f32
        }
    }
    __syncthreads();

    if (tid < BT) {
        pmax[(size_t)(b0 + tid) * NCT + blockIdx.y] = fdec(umax[tid]);
        psum[(size_t)(b0 + tid) * NCT + blockIdx.y] = fsum[tid];
    }
}

// =====================================================================
// Kernel 3: combine per-tile (max,sum) -> logsumexp -> mean NLL
// =====================================================================
__global__ void arcface_finalize(const float* __restrict__ pmax,
                                 const float* __restrict__ psum,
                                 const float* __restrict__ starget,
                                 float* __restrict__ out) {
    const int b = threadIdx.x;   // 512 threads, one per batch row
    float M = -INFINITY, S = 0.0f;
    for (int j = 0; j < NCT; ++j) {
        float m = pmax[(size_t)b * NCT + j];
        float s = psum[(size_t)b * NCT + j];
        if (m > M) { S = S * expf(M - m) + s; M = m; }
        else       { S += s * expf(m - M); }
    }
    float nll = (M + logf(S)) - starget[b];
    __shared__ float red[512];
    red[b] = nll;
    __syncthreads();
    for (int s2 = 256; s2 > 0; s2 >>= 1) {
        if (b < s2) red[b] += red[b + s2];
        __syncthreads();
    }
    if (b == 0) out[0] = red[0] / (float)BATCH;
}

// =====================================================================
extern "C" void kernel_launch(void* const* d_in, const int* in_sizes, int n_in,
                              void* d_out, int out_size, void* d_ws, size_t ws_size,
                              hipStream_t stream) {
    const float* emb    = (const float*)d_in[0];
    const float* weight = (const float*)d_in[1];
    const int*   labels = (const int*)d_in[2];
    float*       out    = (float*)d_out;

    char* ws = (char*)d_ws;
    const size_t OFF_E16  = 0;                                   // 512 KB
    const size_t OFF_TGT  = (size_t)BATCH * EMBED * 2;           // 2 KB
    const size_t OFF_PMAX = OFF_TGT + 2048;                      // 1.6 MB
    const size_t OFF_PSUM = OFF_PMAX + (size_t)BATCH * NCT * 4;  // 1.6 MB

    _Float16* e16     = (_Float16*)(ws + OFF_E16);
    float*    starget = (float*)(ws + OFF_TGT);
    float*    pmax    = (float*)(ws + OFF_PMAX);
    float*    psum    = (float*)(ws + OFF_PSUM);

    arcface_norm_e<<<BATCH, 128, 0, stream>>>(emb, e16);

    dim3 grid(NBT, NCT);   // batch-tile fastest -> adjacent blocks reuse weight tile in L2
    arcface_gemm<<<grid, THREADS, 0, stream>>>(e16, weight, labels, starget, pmax, psum);

    arcface_finalize<<<1, BATCH, 0, stream>>>(pmax, psum, starget, out);
}
